// recurrent_DCRNN_54202487275560
// MI455X (gfx1250) — matrix operand order repaired and checked by reading, not compiled, not run on hardware
//
#include <hip/hip_runtime.h>
#include <hip/hip_bf16.h>

typedef __attribute__((ext_vector_type(16))) _Float16 v16h;
typedef __attribute__((ext_vector_type(8)))  _Float16 v8h;
typedef __attribute__((ext_vector_type(8)))  float    v8f;

#define NNODES 10000
#define F0 256
#define OUTF 128
#define NEDGES 160000

// ---------------- zero init (float4 stores) ----------------
__global__ __launch_bounds__(256) void k_zero(float4* p, int count4) {
    int i = blockIdx.x * blockDim.x + threadIdx.x;
    if (i < count4) p[i] = make_float4(0.f, 0.f, 0.f, 0.f);
}

// ---------------- degrees ----------------
__global__ __launch_bounds__(256) void k_degree(const int* __restrict__ src,
                                                const int* __restrict__ dst,
                                                float* deg_out, float* deg_in, int e) {
    int i = blockIdx.x * blockDim.x + threadIdx.x;
    if (i < e) {
        atomicAdd(&deg_out[src[i]], 1.0f);
        atomicAdd(&deg_in[dst[i]], 1.0f);
    }
}

__global__ __launch_bounds__(256) void k_invert(float* deg, int count) {
    int i = blockIdx.x * blockDim.x + threadIdx.x;
    if (i < count) {
        float v = deg[i];
        deg[i] = (v > 0.f) ? (1.0f / v) : 0.0f;
    }
}

// ---------------- weight prep: f32 -> f16 Bt layouts ----------------
// GEMM B is stored transposed: Bt[n][k], row stride = Ktot halves.
__global__ __launch_bounds__(256) void k_prep_b1(const float* __restrict__ fc0_w,
                                                 _Float16* __restrict__ Bh1) {
    // Bt[n][k] = fc0_w.T[k][n] = fc0_w[n][k]  (identity layout, just cast)
    int i = blockIdx.x * blockDim.x + threadIdx.x;
    if (i < 256 * 256) Bh1[i] = (_Float16)fc0_w[i];
}

__global__ __launch_bounds__(256) void k_prep_b2(const float* __restrict__ wz,
                                                 const float* __restrict__ wh,
                                                 _Float16* __restrict__ Bh2) {
    // Wcat (768 x 256): cols 0:128 = z gate, 128:256 = h gate.
    // rows 0:256   -> W[0,0,c,j] + W[1,0,c,j]   (K=0 identity terms)
    // rows 256:512 -> W[0,1,c,j]                (forward diffusion)
    // rows 512:768 -> W[1,1,c,j]                (reverse diffusion)
    // W flat index: ((d*2 + kk)*384 + c)*128 + j
    int i = blockIdx.x * blockDim.x + threadIdx.x;
    if (i >= 256 * 768) return;
    int n = i / 768;
    int k = i - n * 768;
    const float* W = (n < 128) ? wz : wh;
    int j = (n < 128) ? n : (n - 128);
    float v;
    if (k < 256) {
        v = W[((0) * 384 + k) * 128 + j] + W[((2) * 384 + k) * 128 + j];
    } else if (k < 512) {
        v = W[((1) * 384 + (k - 256)) * 128 + j];
    } else {
        v = W[((3) * 384 + (k - 512)) * 128 + j];
    }
    Bh2[i] = (_Float16)v;
}

__global__ __launch_bounds__(256) void k_prep_bias(const float* __restrict__ bz,
                                                   const float* __restrict__ bh,
                                                   float* __restrict__ bias2) {
    int i = threadIdx.x;
    if (i < 256) bias2[i] = (i < 128) ? bz[i] : bh[i - 128];
}

// ---------------- edge scatter: both diffusion directions ----------------
__global__ __launch_bounds__(256) void k_scatter(const int* __restrict__ src,
                                                 const int* __restrict__ dst,
                                                 const float* __restrict__ inv_out,
                                                 const float* __restrict__ inv_in,
                                                 const float* __restrict__ X,
                                                 float* __restrict__ T1o,
                                                 float* __restrict__ T1i, int e) {
    int tid = threadIdx.x;
    int edge = blockIdx.x * 4 + (tid >> 6);
    int c = (tid & 63) * 4;
    if (edge >= e) return;
    int s = src[edge], d = dst[edge];
    float io = inv_out[s];
    float ii = inv_in[d];
    // forward: T1o[d] += X[s] * inv_out[s]
    float4 xs = *(const float4*)(X + (size_t)s * 256 + c);
    float* po = T1o + (size_t)d * 256 + c;
    atomicAdd(po + 0, xs.x * io);
    atomicAdd(po + 1, xs.y * io);
    atomicAdd(po + 2, xs.z * io);
    atomicAdd(po + 3, xs.w * io);
    // reverse: T1i[s] += X[d] * inv_in[d]
    float4 xd = *(const float4*)(X + (size_t)d * 256 + c);
    float* pi = T1i + (size_t)s * 256 + c;
    atomicAdd(pi + 0, xd.x * ii);
    atomicAdd(pi + 1, xd.y * ii);
    atomicAdd(pi + 2, xd.z * ii);
    atomicAdd(pi + 3, xd.w * ii);
}

// ---------------- WMMA GEMM: D[M,256] = act(A[M,Ktot] * B[Ktot,256] + bias) ----------------
// A is split across up to 3 segments of 256 K each (f32, converted to f16 on stage).
// Bt: f16, transposed [256 n][Ktot k]. 8 waves/block, 64x64 tile, K-step 32.
__global__ __launch_bounds__(256) void k_gemm_wmma(const float* __restrict__ A0,
                                                   const float* __restrict__ A1,
                                                   const float* __restrict__ A2,
                                                   int M, int Ktot,
                                                   const _Float16* __restrict__ Bt,
                                                   const float* __restrict__ bias,
                                                   int act, float* __restrict__ D) {
    __shared__ _Float16 As[64 * 32];
    __shared__ _Float16 Bs[64 * 32];

    const int tid = threadIdx.x;
    const int wave = tid >> 5;
    const int lane = tid & 31;
    const int wm = (wave & 3) * 16;         // wave's M offset in block tile
    const int wnl = (wave >> 2) * 32;       // wave's N offset in block tile
    const int mrow0 = blockIdx.x * 64;
    const int ncol0 = blockIdx.y * 64;

    const int mi = lane & 15;
    const int akb = (lane >> 4) * 16;       // A fragment K sub-base (0 or 16)
    const int bkb = (lane >> 4) * 16;       // B fragment K sub-base (0 or 16)

    // staging coordinates (each thread moves 8 halves)
    const int srow = tid >> 2;              // 0..63
    const int skk = (tid & 3) * 8;          // 0,8,16,24

    v8f acc0 = {};
    v8f acc1 = {};

    for (int k0 = 0; k0 < Ktot; k0 += 32) {
        const float* Aseg;
        int kof;
        if (k0 < 256)      { Aseg = A0; kof = k0; }
        else if (k0 < 512) { Aseg = A1; kof = k0 - 256; }
        else               { Aseg = A2; kof = k0 - 512; }

        // ---- stage A tile (f32 -> f16) ----
        {
            int grow = mrow0 + srow;
            v8h av;
            if (grow < M) {
                const float* sp = Aseg + (size_t)grow * 256 + kof + skk;
                float4 f0 = *(const float4*)(sp);
                float4 f1 = *(const float4*)(sp + 4);
                av[0] = (_Float16)f0.x; av[1] = (_Float16)f0.y;
                av[2] = (_Float16)f0.z; av[3] = (_Float16)f0.w;
                av[4] = (_Float16)f1.x; av[5] = (_Float16)f1.y;
                av[6] = (_Float16)f1.z; av[7] = (_Float16)f1.w;
            } else {
                av = (v8h)(_Float16)0.0f;
            }
            *(v8h*)&As[srow * 32 + skk] = av;
        }
        // ---- stage B tile (already f16, transposed) ----
        {
            const _Float16* sp = Bt + (size_t)(ncol0 + srow) * Ktot + k0 + skk;
            *(v8h*)&Bs[srow * 32 + skk] = *(const v8h*)sp;
        }
        // prefetch next B K-slab through the gfx1250 prefetch path
        if (k0 + 32 < Ktot)
            __builtin_prefetch(Bt + (size_t)(ncol0 + srow) * Ktot + k0 + 32 + skk, 0, 1);

        __syncthreads();

        // ---- A fragment: 16x32, contiguous K order (k = 16h + s) ----
        // lanes 0-15 (M=lane): K = 0..15 ; lanes 16-31: K = 16..31
        v16h a;
        {
            const _Float16* ap = &As[(wm + mi) * 32];
            v8h lo = *(const v8h*)(ap + akb);
            v8h hi = *(const v8h*)(ap + akb + 8);
#pragma unroll
            for (int i = 0; i < 8; i++) { a[i] = lo[i]; a[8 + i] = hi[i]; }
        }
        // ---- B fragments: 32x16, contiguous K order (k = 16h + s) ----
        // lanes 0-15 (N=lane): K = 0..15 ; lanes 16-31: K = 16..31
        v16h b0, b1;
        {
            const _Float16* bp0 = &Bs[(wnl + mi) * 32 + bkb];
            v8h l0 = *(const v8h*)(bp0);
            v8h h0 = *(const v8h*)(bp0 + 8);
            const _Float16* bp1 = &Bs[(wnl + 16 + mi) * 32 + bkb];
            v8h l1 = *(const v8h*)(bp1);
            v8h h1 = *(const v8h*)(bp1 + 8);
#pragma unroll
            for (int i = 0; i < 8; i++) {
                b0[i] = l0[i]; b0[8 + i] = h0[i];
                b1[i] = l1[i]; b1[8 + i] = h1[i];
            }
        }

        acc0 = __builtin_amdgcn_wmma_f32_16x16x32_f16(false, a, false, b0,
                                                      (short)0, acc0, false, false);
        acc1 = __builtin_amdgcn_wmma_f32_16x16x32_f16(false, a, false, b1,
                                                      (short)0, acc1, false, false);
        __syncthreads();
    }

    // ---- store with bias (+ optional ReLU). C/D layout: VGPR r -> M = r (+8 for hi lanes) ----
    const int rbase = (lane < 16) ? 0 : 8;
    const int gc0 = ncol0 + wnl + (lane & 15);
    const int gc1 = gc0 + 16;
    const float bia0 = bias[gc0];
    const float bia1 = bias[gc1];
#pragma unroll
    for (int r = 0; r < 8; r++) {
        int grow = mrow0 + wm + rbase + r;
        if (grow < M) {
            float v0 = acc0[r] + bia0;
            float v1 = acc1[r] + bia1;
            if (act) { v0 = fmaxf(v0, 0.f); v1 = fmaxf(v1, 0.f); }
            D[(size_t)grow * 256 + gc0] = v0;
            D[(size_t)grow * 256 + gc1] = v1;
        }
    }
}

// ---------------- epilogue: gates + H + final fc ----------------
__global__ __launch_bounds__(128) void k_epilogue(const float* __restrict__ G,
                                                  const float* __restrict__ fc_w,
                                                  const float* __restrict__ fc_b,
                                                  float* __restrict__ out,
                                                  float* __restrict__ H) {
    __shared__ float red[128];
    int n = blockIdx.x;
    int f = threadIdx.x;
    float zp = G[(size_t)n * 256 + f];
    float hp = G[(size_t)n * 256 + 128 + f];
    float z = 1.0f / (1.0f + expf(-zp));
    float ht = tanhf(hp);
    float h = (1.0f - z) * ht;               // H_prev = 0  =>  H = (1-Z)*H_tilde
    H[(size_t)n * 128 + f] = h;
    red[f] = fmaxf(h, 0.f) * fc_w[f];
    __syncthreads();
#pragma unroll
    for (int s = 64; s > 0; s >>= 1) {
        if (f < s) red[f] += red[f + s];
        __syncthreads();
    }
    if (f == 0) out[n] = red[0] + fc_b[0];
}

// ---------------- launcher ----------------
extern "C" void kernel_launch(void* const* d_in, const int* in_sizes, int n_in,
                              void* d_out, int out_size, void* d_ws, size_t ws_size,
                              hipStream_t stream) {
    (void)in_sizes; (void)n_in; (void)out_size; (void)ws_size;

    const float* x     = (const float*)d_in[0];
    const int*   ei    = (const int*)d_in[1];
    const float* fc0_w = (const float*)d_in[2];
    const float* fc0_b = (const float*)d_in[3];
    const float* wz    = (const float*)d_in[4];
    const float* bz    = (const float*)d_in[5];
    // d_in[6]=wr, d_in[7]=br are dead (H_prev==0 makes the R gate unused)
    const float* wh    = (const float*)d_in[8];
    const float* bh    = (const float*)d_in[9];
    const float* fc_w  = (const float*)d_in[10];
    const float* fc_b  = (const float*)d_in[11];

    const int* src = ei;
    const int* dst = ei + NEDGES;

    // workspace carve (256B aligned slots)
    char* w = (char*)d_ws;
    size_t off = 0;
    auto carve = [&](size_t bytes) -> char* {
        char* p = w + off;
        off += (bytes + 255) & ~(size_t)255;
        return p;
    };
    float* deg   = (float*)carve(2 * NNODES * sizeof(float));   // deg_out | deg_in
    float* T1o   = (float*)carve((size_t)NNODES * 256 * sizeof(float));
    float* T1i   = (float*)carve((size_t)NNODES * 256 * sizeof(float));
    float* X     = (float*)carve((size_t)NNODES * 256 * sizeof(float));
    float* G     = (float*)carve((size_t)NNODES * 256 * sizeof(float));
    float* bias2 = (float*)carve(256 * sizeof(float));
    _Float16* Bh1 = (_Float16*)carve(256 * 256 * sizeof(_Float16));
    _Float16* Bh2 = (_Float16*)carve(256 * 768 * sizeof(_Float16));

    float* deg_out = deg;
    float* deg_in  = deg + NNODES;
    float* out_v = (float*)d_out;          // [N]
    float* H_v   = out_v + NNODES;         // [N,128]

    // 1) zero deg + T1o + T1i (contiguous region thanks to carve order & 256B pads)
    {
        // zero each region separately to be safe w.r.t. alignment padding
        int c4a = (2 * NNODES) / 4;
        k_zero<<<(c4a + 255) / 256, 256, 0, stream>>>((float4*)deg, c4a);
        int c4b = (NNODES * 256) / 4;
        k_zero<<<(c4b + 255) / 256, 256, 0, stream>>>((float4*)T1o, c4b);
        k_zero<<<(c4b + 255) / 256, 256, 0, stream>>>((float4*)T1i, c4b);
    }

    // 2) degrees -> inverse degrees
    k_degree<<<(NEDGES + 255) / 256, 256, 0, stream>>>(src, dst, deg_out, deg_in, NEDGES);
    k_invert<<<(2 * NNODES + 255) / 256, 256, 0, stream>>>(deg, 2 * NNODES);

    // 3) weight / bias prep (f32 -> f16 Bt layouts)
    k_prep_b1<<<256, 256, 0, stream>>>(fc0_w, Bh1);
    k_prep_b2<<<(256 * 768 + 255) / 256, 256, 0, stream>>>(wz, wh, Bh2);
    k_prep_bias<<<1, 256, 0, stream>>>(bz, bh, bias2);

    // 4) X = relu(x @ fc0_w.T + fc0_b)
    {
        dim3 grid((NNODES + 63) / 64, 4);
        k_gemm_wmma<<<grid, 256, 0, stream>>>(x, nullptr, nullptr, NNODES, 256,
                                              Bh1, fc0_b, 1, X);
    }

    // 5) diffusion scatter: T1o[dst]+=X[src]*inv_out[src], T1i[src]+=X[dst]*inv_in[dst]
    k_scatter<<<NEDGES / 4, 256, 0, stream>>>(src, dst, deg_out, deg_in, X, T1o, T1i, NEDGES);

    // 6) G = [X | T1o | T1i] @ Wcat + [bz|bh]   (z pre-act in cols 0:128, h in 128:256)
    {
        dim3 grid((NNODES + 63) / 64, 4);
        k_gemm_wmma<<<grid, 256, 0, stream>>>(X, T1o, T1i, NNODES, 768,
                                              Bh2, bias2, 0, G);
    }

    // 7) H = (1 - sigmoid(z)) * tanh(h);  out = relu(H) @ fc_w.T + fc_b
    k_epilogue<<<NNODES, 128, 0, stream>>>(G, fc_w, fc_b, out_v, H_v);
}
